// BasicBlock_17575006175305
// MI455X (gfx1250) — 8-bit matrix operand order repaired and checked by reading, compiled, not run on hardware
//
#include <hip/hip_runtime.h>
#include <stdint.h>
#include <stddef.h>


typedef __attribute__((ext_vector_type(8))) int v8i;

#define CIN    64
#define COUT   128
#define HD     112
#define WD     112
#define HO     56
#define WO     56
#define BATCH  32
#define PIX    (HO * WO)

#define OUT1_OFF   ((size_t)0)
#define OUT1_BYTES ((size_t)BATCH * CIN * PIX * 4)
#define BIN1_OFF   (OUT1_OFF + OUT1_BYTES)
#define BIN1_BYTES ((size_t)BATCH * PIX * CIN)
#define CONVB_OFF  (BIN1_OFF + BIN1_BYTES)
#define CONVB_BYTES ((size_t)36 * 1024)
#define PW1B_OFF   (CONVB_OFF + CONVB_BYTES)
#define PW2B_OFF   (PW1B_OFF + 4096)
#define SCALE_OFF  (PW2B_OFF + 4096)

static __device__ __forceinline__ char bsign(float v) {
    return (v >= 0.f) ? (char)1 : (char)-1;
}

__global__ __launch_bounds__(256) void reduce_scales_kernel(
    const float* __restrict__ w3, const float* __restrict__ wpw1,
    const float* __restrict__ wpw2, float* __restrict__ scales)
{
    __shared__ float red[256];
    const float* src;
    int n;
    if (blockIdx.x == 0)      { src = w3;   n = CIN * CIN * 9; }
    else if (blockIdx.x == 1) { src = wpw1; n = CIN * CIN; }
    else                      { src = wpw2; n = CIN * CIN; }
    float s = 0.f;
    for (int i = threadIdx.x; i < n; i += 256) s += fabsf(src[i]);
    red[threadIdx.x] = s;
    __syncthreads();
    for (int off = 128; off > 0; off >>= 1) {
        if ((int)threadIdx.x < off) red[threadIdx.x] += red[threadIdx.x + off];
        __syncthreads();
    }
    if (threadIdx.x == 0) scales[blockIdx.x] = red[0] / (float)n;
}

__global__ __launch_bounds__(256) void pack_weights_kernel(
    const float* __restrict__ w3, const float* __restrict__ wpw1,
    const float* __restrict__ wpw2, char* __restrict__ convB,
    char* __restrict__ pw1B, char* __restrict__ pw2B)
{
    const int stride = blockDim.x * gridDim.x;
    const int tid = blockIdx.x * blockDim.x + threadIdx.x;

    for (int idx = tid; idx < 36 * 1024; idx += stride) {
        int frag = idx >> 10, rem = idx & 1023;
        int lane = rem >> 5, v = (rem >> 2) & 7, j = rem & 3;
        int ks = frag >> 2, nt = frag & 3;
        int n = lane & 15, hi = lane >> 4;
        int k = ((v >> 1) << 4) + (hi << 3) + ((v & 1) << 2) + j;
        int cout = nt * 16 + n;
        int ky = ks / 3, kx = ks % 3;
        convB[idx] = bsign(w3[((cout * CIN + k) * 3 + ky) * 3 + kx]);
    }
    for (int idx = tid; idx < 4 * 1024; idx += stride) {
        int frag = idx >> 10, rem = idx & 1023;
        int lane = rem >> 5, v = (rem >> 2) & 7, j = rem & 3;
        int n = lane & 15, hi = lane >> 4;
        int k = ((v >> 1) << 4) + (hi << 3) + ((v & 1) << 2) + j;
        int cout = frag * 16 + n;
        pw1B[idx] = bsign(wpw1[cout * CIN + k]);
        pw2B[idx] = bsign(wpw2[cout * CIN + k]);
    }
}

__global__ __launch_bounds__(256) void stage1_kernel(
    const float* __restrict__ x,   const float* __restrict__ b11,
    const float* __restrict__ b12, const float* __restrict__ b13,
    const float* __restrict__ b21, const float* __restrict__ a1,
    const char*  __restrict__ convB, const float* __restrict__ scales,
    float* __restrict__ out1, char* __restrict__ bin1)
{
    __shared__ signed char binA[17 * 17 * 64];

    const int b = blockIdx.y;
    const int oy0 = (blockIdx.x / 7) * 8;
    const int ox0 = (blockIdx.x % 7) * 8;
    const int iy0 = 2 * oy0 - 1, ix0 = 2 * ox0 - 1;
    const int tid = threadIdx.x;

    for (int idx = tid; idx < 17 * 17 * 16; idx += 256) {
        int pos = idx >> 4, cg = idx & 15;
        int ly = pos / 17, lx = pos % 17;
        int gy = iy0 + ly, gx = ix0 + lx;
        uint32_t packed = 0;
        if (gy >= 0 && gy < HD && gx >= 0 && gx < WD) {
            int c0 = cg << 2;
            #pragma unroll
            for (int j = 0; j < 4; j++) {
                int c = c0 + j;
                float v = x[((size_t)(b * CIN + c) * HD + gy) * WD + gx] + b11[c];
                packed |= (uint32_t)((v >= 0.f) ? 0x01u : 0xFFu) << (8 * j);
            }
        }
        *(uint32_t*)(&binA[(pos << 6) + (cg << 2)]) = packed;
    }
    __syncthreads();

    const int wave = tid >> 5, lane = tid & 31;
    const int n = lane & 15, hi = lane >> 4;
    const int am = n;
    const int nt = wave & 3;
    const int mtBase = wave >> 2;

    const int sy0 = ((mtBase) >> 1) * 4,     sx0 = ((mtBase) & 1) * 4;
    const int sy1 = ((mtBase + 2) >> 1) * 4, sx1 = ((mtBase + 2) & 1) * 4;
    const int ady = am >> 2, adx = am & 3;

    v8i acc0 = {}; v8i acc1 = {};

    #pragma unroll
    for (int ks = 0; ks < 9; ks++) {
        const int ky = ks / 3, kx = ks % 3;
        const int* bp = (const int*)(convB + (size_t)(ks * 4 + nt) * 1024) + lane * 8;
        v8i bf;
        #pragma unroll
        for (int v = 0; v < 8; v++) bf[v] = bp[v];

        {
            const signed char* arow =
                binA + (((2 * (sy0 + ady) + ky) * 17 + (2 * (sx0 + adx) + kx)) << 6);
            v8i af;
            #pragma unroll
            for (int v = 0; v < 8; v++) {
                int chunk = ((v >> 1) << 2) + (hi << 1) + (v & 1);
                af[v] = *(const int*)(arow + (chunk << 2));
            }
            acc0 = __builtin_amdgcn_wmma_i32_16x16x64_iu8(true, af, true, bf, acc0, false, false);
        }
        {
            const signed char* arow =
                binA + (((2 * (sy1 + ady) + ky) * 17 + (2 * (sx1 + adx) + kx)) << 6);
            v8i af;
            #pragma unroll
            for (int v = 0; v < 8; v++) {
                int chunk = ((v >> 1) << 2) + (hi << 1) + (v & 1);
                af[v] = *(const int*)(arow + (chunk << 2));
            }
            acc1 = __builtin_amdgcn_wmma_i32_16x16x64_iu8(true, af, true, bf, acc1, false, false);
        }
    }

    const float scale3 = scales[0];
    const int cout = nt * 16 + n;
    const float bias12 = b12[cout], bias13 = b13[cout];
    const float alpha = a1[cout],  bias21 = b21[cout];

    #pragma unroll
    for (int half = 0; half < 2; half++) {
        const int sy = half ? sy1 : sy0;
        const int sx = half ? sx1 : sx0;
        #pragma unroll
        for (int v = 0; v < 8; v++) {
            const int M = v + 8 * hi;
            const int dy = M >> 2, dx = M & 3;
            const int oy = oy0 + sy + dy, ox = ox0 + sx + dx;
            const float cv = (float)(half ? acc1[v] : acc0[v]) * scale3;
            const float* xp = &x[((size_t)(b * CIN + cout) * HD + 2 * oy) * WD + 2 * ox];
            const float sc = 0.25f * (xp[0] + xp[1] + xp[WD] + xp[WD + 1]);
            float t = cv + sc + bias12;
            t = (t >= 0.f) ? t : alpha * t;
            t += bias13;
            const size_t opix = (size_t)oy * WO + ox;
            out1[(size_t)(b * CIN + cout) * PIX + opix] = t;
            bin1[((size_t)b * PIX + opix) * CIN + cout] = bsign(t + bias21);
        }
    }
}

__global__ __launch_bounds__(256) void stage2_kernel(
    const char*  __restrict__ bin1, const char* __restrict__ pw1B,
    const char*  __restrict__ pw2B, const float* __restrict__ out1,
    const float* __restrict__ scales, const float* __restrict__ b22,
    const float* __restrict__ b23, const float* __restrict__ a2,
    float* __restrict__ dout)
{
    const int tid = threadIdx.x;
    const int wave = tid >> 5, lane = tid & 31;
    const int gw = blockIdx.x * 8 + wave;
    const int b = gw / (196 * 4);
    const int rem = gw % (196 * 4);
    const int mt = rem >> 2, nt = rem & 3;
    const int n = lane & 15, hi = lane >> 4;
    const int pix0 = mt * 16;

    const char* abase = bin1 + ((size_t)b * PIX + pix0 + n) * CIN;
    v8i af;
    #pragma unroll
    for (int v = 0; v < 8; v++) {
        int chunk = ((v >> 1) << 2) + (hi << 1) + (v & 1);
        af[v] = *(const int*)(abase + (chunk << 2));
    }

    const int* b1p = (const int*)(pw1B + (size_t)nt * 1024) + lane * 8;
    const int* b2p = (const int*)(pw2B + (size_t)nt * 1024) + lane * 8;
    v8i bf1, bf2;
    #pragma unroll
    for (int v = 0; v < 8; v++) { bf1[v] = b1p[v]; bf2[v] = b2p[v]; }

    const v8i zero = {};
    v8i acc1 = __builtin_amdgcn_wmma_i32_16x16x64_iu8(true, af, true, bf1, zero, false, false);
    v8i acc2 = __builtin_amdgcn_wmma_i32_16x16x64_iu8(true, af, true, bf2, zero, false, false);

    const float s1 = scales[1], s2 = scales[2];
    const int c1 = nt * 16 + n, c2 = c1 + CIN;
    const float bb22a = b22[c1], bb23a = b23[c1], al1 = a2[c1];
    const float bb22b = b22[c2], bb23b = b23[c2], al2 = a2[c2];

    #pragma unroll
    for (int v = 0; v < 8; v++) {
        const int M = v + 8 * hi;
        const size_t pix = (size_t)pix0 + M;
        const float res = out1[(size_t)(b * CIN + c1) * PIX + pix];
        float t1 = (float)acc1[v] * s1 + res + bb22a;
        t1 = (t1 >= 0.f) ? t1 : al1 * t1;
        dout[(size_t)(b * COUT + c1) * PIX + pix] = t1 + bb23a;
        float t2 = (float)acc2[v] * s2 + res + bb22b;
        t2 = (t2 >= 0.f) ? t2 : al2 * t2;
        dout[(size_t)(b * COUT + c2) * PIX + pix] = t2 + bb23b;
    }
}

extern "C" void kernel_launch(void* const* d_in, const int* in_sizes, int n_in,
                              void* d_out, int out_size, void* d_ws, size_t ws_size,
                              hipStream_t stream)
{
    (void)in_sizes; (void)n_in; (void)out_size; (void)ws_size;
    const float* x    = (const float*)d_in[0];
    const float* b11  = (const float*)d_in[1];
    const float* b12  = (const float*)d_in[2];
    const float* b13  = (const float*)d_in[3];
    const float* b21  = (const float*)d_in[4];
    const float* b22  = (const float*)d_in[5];
    const float* b23  = (const float*)d_in[6];
    const float* w3   = (const float*)d_in[7];
    const float* wpw1 = (const float*)d_in[8];
    const float* wpw2 = (const float*)d_in[9];
    const float* a1   = (const float*)d_in[10];
    const float* a2   = (const float*)d_in[11];

    char*  ws     = (char*)d_ws;
    float* out1   = (float*)(ws + OUT1_OFF);
    char*  bin1   = ws + BIN1_OFF;
    char*  convB  = ws + CONVB_OFF;
    char*  pw1B   = ws + PW1B_OFF;
    char*  pw2B   = ws + PW2B_OFF;
    float* scales = (float*)(ws + SCALE_OFF);

    reduce_scales_kernel<<<3, 256, 0, stream>>>(w3, wpw1, wpw2, scales);
    pack_weights_kernel<<<64, 256, 0, stream>>>(w3, wpw1, wpw2, convB, pw1B, pw2B);
    stage1_kernel<<<dim3(49, BATCH), 256, 0, stream>>>(
        x, b11, b12, b13, b21, a1, convB, scales, out1, bin1);
    stage2_kernel<<<(BATCH * 196 * 4) / 8, 256, 0, stream>>>(
        bin1, pw1B, pw2B, out1, scales, b22, b23, a2, (float*)d_out);
}
